// MurderTreeV2Core_21672404976194
// MI455X (gfx1250) — compile-verified, matrix operand order checked by reading, not run on hardware
//
#include <hip/hip_runtime.h>
#include <hip/hip_bf16.h>

// Griffin block for MI455X (gfx1250, wave32, WMMA + TDM).
// Compute-bound (~172 GFLOP vs ~10us of HBM time at 23.3 TB/s): all GEMMs and
// attention matmuls go through v_wmma_f32_16x16x32_bf16. fp32 weights are
// converted to bf16 during the global->LDS tile stage (VALU path), while the
// pure-bf16 tiles (GEMM A, attention Q/K) are moved by the Tensor Data Mover
// (tensor_load_to_lds + s_wait_tensorcnt).

typedef __bf16 bf16_t;
typedef bf16_t v16bf __attribute__((ext_vector_type(16)));
typedef float  v8f   __attribute__((ext_vector_type(8)));
typedef unsigned int v4u __attribute__((ext_vector_type(4)));
typedef int v4i __attribute__((ext_vector_type(4)));
typedef int v8i __attribute__((ext_vector_type(8)));

#define GB 2
#define GS 2048
#define GD 1024
#define GH 16
#define GHD 64
#define GE 4096
#define GM (GB * GS)
#define MASK_VAL (-1e4f)

#if __has_builtin(__builtin_amdgcn_tensor_load_to_lds) && \
    __has_builtin(__builtin_amdgcn_s_wait_tensorcnt)
#define HAVE_TDM 1
#else
#define HAVE_TDM 0
#endif

union PackU {
    struct { uint4 a, b; } s;
    v16bf v;
};

static __device__ inline v16bf pack_bf16(uint4 lo, uint4 hi) {
    PackU u;
    u.s.a = lo;
    u.s.b = hi;
    return u.v;
}

#if HAVE_TDM
// 2-D TDM load: tile_h rows of tile_w bf16 elements, global row stride
// stride_elems, into LDS at lds_off with per-row padding (pad fields encode
// the LDS row stride; see CDNA5 ISA).
static __device__ inline void tdm_load_2d_bf16(const void* gaddr, unsigned int lds_off,
                                               unsigned int tile_w, unsigned int tile_h,
                                               unsigned long long stride_elems,
                                               unsigned int pad_interval,
                                               unsigned int pad_amount) {
    unsigned long long ga = (unsigned long long)(size_t)gaddr;
    v4u g0;
    g0.x = 1u;  // count=1, is_restore=0, gather off
    g0.y = lds_off;
    g0.z = (unsigned int)(ga & 0xffffffffu);
    g0.w = (unsigned int)((ga >> 32) & 0x01ffffffu) | (2u << 30);  // type=2 (image)
    v8i g1;
    // w0: workgroup_mask=0, data_size=1 (2B), pad_enable, pad_interval, pad_amount
    g1[0] = (int)((1u << 16) | (1u << 20) | (pad_interval << 22) | (pad_amount << 25));
    g1[1] = (int)((tile_w & 0xffffu) << 16);  // abar_addr=0, tensor_dim0 lo16
    g1[2] = (int)(((tile_w >> 16) & 0xffffu) | ((tile_h & 0xffffu) << 16));  // dim0 hi, dim1 lo
    g1[3] = (int)(((tile_h >> 16) & 0xffffu) | ((tile_w & 0xffffu) << 16));  // dim1 hi, tile_dim0
    g1[4] = (int)(tile_h & 0xffffu);  // tile_dim1 (tile_dim2=0)
    g1[5] = (int)(unsigned int)(stride_elems & 0xffffffffu);  // tensor_dim0_stride lo
    g1[6] = (int)(unsigned int)((stride_elems >> 32) & 0xffffu);  // stride hi, dim1_stride=0
    g1[7] = 0;
    v4i z4 = {0, 0, 0, 0};
#if defined(__clang_major__) && (__clang_major__ >= 23)
    v8i z8 = {0, 0, 0, 0, 0, 0, 0, 0};
    __builtin_amdgcn_tensor_load_to_lds(g0, g1, z4, z4, z8, 0);
#else
    __builtin_amdgcn_tensor_load_to_lds(g0, g1, z4, z4, 0);
#endif
}
#endif

// ---------------------------------------------------------------------------
// RMSNorm (fp32 in) -> bf16 out
// ---------------------------------------------------------------------------
__global__ __launch_bounds__(256) void rmsnorm_cast_kernel(
    const float* __restrict__ x, const float* __restrict__ w,
    bf16_t* __restrict__ out) {
    const int t = blockIdx.x;
    const float* row = x + (size_t)t * GD;
    float ss = 0.f;
    for (int c = threadIdx.x; c < GD; c += 256) {
        float v = row[c];
        ss += v * v;
    }
    __shared__ float red[256];
    red[threadIdx.x] = ss;
    __syncthreads();
    for (int s = 128; s > 0; s >>= 1) {
        if (threadIdx.x < s) red[threadIdx.x] += red[threadIdx.x + s];
        __syncthreads();
    }
    const float rs = rsqrtf(red[0] / (float)GD + 1e-5f);
    for (int c = threadIdx.x; c < GD; c += 256) {
        out[(size_t)t * GD + c] = (bf16_t)(row[c] * rs * w[c]);
    }
}

// ---------------------------------------------------------------------------
// Generic GEMM: C[M,N] = A[M,K] (bf16) x W[N,K]^T (fp32 -> bf16 on the fly)
// Block tile 128x128, 8 waves, each wave 64(M)x32(N) = 8 16x16 frags.
// A tile arrives via TDM (tensor_load_to_lds); W tile converted via VALU.
// MODE 0: out_bf = bf16(acc [+ bias])
// MODE 1: out_f  = res + acc
// MODE 2: out_f  = res + sigmoid(gate[col]) * (acc + db[col])
// ---------------------------------------------------------------------------
template <int MODE>
__global__ __launch_bounds__(256) void gemm_bf16_kernel(
    const bf16_t* __restrict__ A, const float* __restrict__ W,
    int M, int N, int K,
    bf16_t* __restrict__ out_bf, const float* __restrict__ bias,
    float* __restrict__ out_f, const float* __restrict__ res,
    const float* __restrict__ gate, const float* __restrict__ db) {
    __shared__ bf16_t a_lds[128 * 48];
    __shared__ bf16_t b_lds[128 * 48];

    const int tid = threadIdx.x;
    const int wv = tid >> 5;
    const int lane = tid & 31;
    const int half = lane >> 4;
    const int r = lane & 15;
    const int waveM = wv >> 2;  // 0..1  -> 64 rows each
    const int waveN = wv & 3;   // 0..3  -> 32 cols each
    const int mBase = blockIdx.y * 128;
    const int nBase = blockIdx.x * 128;

    v8f acc[4][2];
#pragma unroll
    for (int i = 0; i < 4; ++i)
#pragma unroll
        for (int j = 0; j < 2; ++j) acc[i][j] = (v8f)0.f;

    const int nkt = K / 32;
    for (int kt = 0; kt < nkt; ++kt) {
        // A tile: 128 rows x 32 bf16, LDS row stride 48 bf16 (16 DW + 8 DW pad)
#if HAVE_TDM
        if (wv == 0) {
            tdm_load_2d_bf16(A + (size_t)mBase * K + (size_t)kt * 32,
                             (unsigned int)(size_t)a_lds,
                             32u, 128u, (unsigned long long)K,
                             /*pad_interval=*/3u /*16 DW*/, /*pad_amount=*/7u /*8 DW*/);
        }
#else
#pragma unroll
        for (int i = 0; i < 2; ++i) {
            int idx = tid + i * 256;
            int row = idx >> 2;
            int ch = idx & 3;
            *(uint4*)&a_lds[row * 48 + ch * 8] =
                *(const uint4*)(A + (size_t)(mBase + row) * K + kt * 32 + ch * 8);
        }
#endif
        // W tile: 128 rows x 32 fp32 -> bf16
        {
            int row = tid >> 1;
            int hf = tid & 1;
            const float4* src =
                (const float4*)(W + (size_t)(nBase + row) * K + kt * 32 + hf * 16);
            float fv[16];
            *(float4*)&fv[0] = src[0];
            *(float4*)&fv[4] = src[1];
            *(float4*)&fv[8] = src[2];
            *(float4*)&fv[12] = src[3];
            bf16_t tmp[16];
#pragma unroll
            for (int j = 0; j < 16; ++j) tmp[j] = (bf16_t)fv[j];
            *(uint4*)&b_lds[row * 48 + hf * 16] = *(uint4*)&tmp[0];
            *(uint4*)&b_lds[row * 48 + hf * 16 + 8] = *(uint4*)&tmp[8];
        }
        if (kt + 1 < nkt) {  // next W tile prefetch -> global_prefetch_b8
            __builtin_prefetch(W + (size_t)(nBase + (tid >> 1)) * K + (kt + 1) * 32 + (tid & 1) * 16, 0, 1);
        }
#if HAVE_TDM
        if (wv == 0) __builtin_amdgcn_s_wait_tensorcnt(0);
#endif
        __syncthreads();

        v16bf af[4], bfr[2];
#pragma unroll
        for (int fi = 0; fi < 4; ++fi) {
            int row = waveM * 64 + fi * 16 + r;
            af[fi] = pack_bf16(*(uint4*)&a_lds[row * 48 + half * 8],
                               *(uint4*)&a_lds[row * 48 + 16 + half * 8]);
        }
#pragma unroll
        for (int fj = 0; fj < 2; ++fj) {
            int row = waveN * 32 + fj * 16 + r;
            bfr[fj] = pack_bf16(*(uint4*)&b_lds[row * 48 + half * 8],
                                *(uint4*)&b_lds[row * 48 + 16 + half * 8]);
        }
#pragma unroll
        for (int fi = 0; fi < 4; ++fi)
#pragma unroll
            for (int fj = 0; fj < 2; ++fj)
                acc[fi][fj] = __builtin_amdgcn_wmma_f32_16x16x32_bf16(
                    false, af[fi], false, bfr[fj], (short)0, acc[fi][fj], false, false);
        __syncthreads();
    }

    // Epilogue. C/D layout: lane holds col (lane&15), rows v + 8*(lane>>4).
#pragma unroll
    for (int fi = 0; fi < 4; ++fi)
#pragma unroll
        for (int fj = 0; fj < 2; ++fj)
#pragma unroll
            for (int vv = 0; vv < 8; ++vv) {
                int row = mBase + waveM * 64 + fi * 16 + 8 * half + vv;
                int col = nBase + waveN * 32 + fj * 16 + r;
                float val = acc[fi][fj][vv];
                size_t off = (size_t)row * N + col;
                if constexpr (MODE == 0) {
                    if (bias) val += bias[col];
                    out_bf[off] = (bf16_t)val;
                } else if constexpr (MODE == 1) {
                    out_f[off] = res[off] + val;
                } else {
                    val += db[col];
                    float g = 1.f / (1.f + expf(-gate[col]));
                    out_f[off] = res[off] + g * val;
                }
            }
}

// ---------------------------------------------------------------------------
// RoPE on q,k (in place, bf16). cos/sin tables [S, 64], cos[p,i]==cos[p,i+32].
// ---------------------------------------------------------------------------
__global__ __launch_bounds__(256) void rope_kernel(
    bf16_t* __restrict__ q, bf16_t* __restrict__ k,
    const float* __restrict__ cosT, const float* __restrict__ sinT) {
    int idx = blockIdx.x * 256 + threadIdx.x;  // M * H * 32 threads
    int t = idx >> 9;
    int rem = idx & 511;
    int head = rem >> 5;
    int i = rem & 31;
    int pos = t & (GS - 1);
    size_t base = (size_t)t * GD + head * GHD;
    float c1 = cosT[pos * GHD + i];
    float s1 = sinT[pos * GHD + i];
    float c2 = cosT[pos * GHD + i + 32];
    float s2 = sinT[pos * GHD + i + 32];
    {
        float x1 = (float)q[base + i], x2 = (float)q[base + i + 32];
        q[base + i] = (bf16_t)(x1 * c1 - x2 * s1);
        q[base + i + 32] = (bf16_t)(x2 * c2 + x1 * s2);
    }
    {
        float x1 = (float)k[base + i], x2 = (float)k[base + i + 32];
        k[base + i] = (bf16_t)(x1 * c1 - x2 * s1);
        k[base + i + 32] = (bf16_t)(x2 * c2 + x1 * s2);
    }
}

// ---------------------------------------------------------------------------
// Flash attention: WG = (qb, h, b); 128 q rows (16 per wave), kv blocks of 64.
// Q/K tiles via TDM; V transposed into LDS by VALU. Scores and P*V via WMMA.
// ---------------------------------------------------------------------------
__global__ __launch_bounds__(256) void attn_kernel(
    const bf16_t* __restrict__ q, const bf16_t* __restrict__ k,
    const bf16_t* __restrict__ v, bf16_t* __restrict__ o) {
    const int qb = blockIdx.x;
    const int h = blockIdx.y;
    const int b = blockIdx.z;

    __shared__ bf16_t qs[128 * 72];
    __shared__ bf16_t ks[64 * 72];
    __shared__ bf16_t vts[64 * 72];  // V transposed: [hd][kv]
    __shared__ bf16_t ps[128 * 72];

    const int tid = threadIdx.x;
    const int wv = tid >> 5;
    const int lane = tid & 31;
    const int half = lane >> 4;
    const int r = lane & 15;
    const int tokBase = b * GS + qb * 128;

    // Q tile: 128 x 64 bf16, LDS row stride 72 bf16 (32 DW + 4 DW pad)
#if HAVE_TDM
    if (wv == 0) {
        tdm_load_2d_bf16(q + (size_t)tokBase * GD + h * GHD,
                         (unsigned int)(size_t)qs, 64u, 128u,
                         (unsigned long long)GD,
                         /*pad_interval=*/4u /*32 DW*/, /*pad_amount=*/3u /*4 DW*/);
        __builtin_amdgcn_s_wait_tensorcnt(0);
    }
#else
#pragma unroll
    for (int i = 0; i < 4; ++i) {
        int idx = tid + i * 256;
        int row = idx >> 3;
        int ch = idx & 7;
        *(uint4*)&qs[row * 72 + ch * 8] =
            *(const uint4*)(q + (size_t)(tokBase + row) * GD + h * GHD + ch * 8);
    }
#endif
    __syncthreads();

    v16bf aq[2];
#pragma unroll
    for (int kk = 0; kk < 2; ++kk) {
        int row = wv * 16 + r;
        aq[kk] = pack_bf16(*(uint4*)&qs[row * 72 + kk * 32 + half * 8],
                           *(uint4*)&qs[row * 72 + kk * 32 + 16 + half * 8]);
    }

    float mstat[8], lstat[8];
    v8f oacc[4];
#pragma unroll
    for (int i = 0; i < 8; ++i) {
        mstat[i] = -3.0e38f;
        lstat[i] = 0.f;
    }
#pragma unroll
    for (int j = 0; j < 4; ++j) oacc[j] = (v8f)0.f;

    const float scale = 0.125f;  // 1/sqrt(64)
    const int nkb = (qb + 1) * 2;
    for (int kb = 0; kb < nkb; ++kb) {
        // K tile 64x64 via TDM; V tile transposed into vts by VALU/DS.
#if HAVE_TDM
        if (wv == 0) {
            tdm_load_2d_bf16(k + (size_t)(b * GS + kb * 64) * GD + h * GHD,
                             (unsigned int)(size_t)ks, 64u, 64u,
                             (unsigned long long)GD, 4u, 3u);
        }
#endif
#pragma unroll
        for (int i = 0; i < 2; ++i) {
            int idx = tid + i * 256;
            int row = idx >> 3;
            int ch = idx & 7;
#if !HAVE_TDM
            *(uint4*)&ks[row * 72 + ch * 8] =
                *(const uint4*)(k + (size_t)(b * GS + kb * 64 + row) * GD + h * GHD + ch * 8);
#endif
            uint4 t4 = *(const uint4*)(v + (size_t)(b * GS + kb * 64 + row) * GD + h * GHD + ch * 8);
            const bf16_t* e = (const bf16_t*)&t4;
#pragma unroll
            for (int j = 0; j < 8; ++j) vts[(ch * 8 + j) * 72 + row] = e[j];
        }
#if HAVE_TDM
        if (wv == 0) __builtin_amdgcn_s_wait_tensorcnt(0);
#endif
        __syncthreads();

        // Scores: S = Q(16x64) x K^T(64x64) per wave -> 4 frags
        v8f sc[4];
#pragma unroll
        for (int j = 0; j < 4; ++j) sc[j] = (v8f)0.f;
#pragma unroll
        for (int kk = 0; kk < 2; ++kk) {
#pragma unroll
            for (int j = 0; j < 4; ++j) {
                int row = j * 16 + r;
                v16bf bk = pack_bf16(*(uint4*)&ks[row * 72 + kk * 32 + half * 8],
                                     *(uint4*)&ks[row * 72 + kk * 32 + 16 + half * 8]);
                sc[j] = __builtin_amdgcn_wmma_f32_16x16x32_bf16(
                    false, aq[kk], false, bk, (short)0, sc[j], false, false);
            }
        }

        // Scale + causal mask + online softmax (row stats per v, 16-lane reduce)
#pragma unroll
        for (int vv = 0; vv < 8; ++vv) {
            int qrow = qb * 128 + wv * 16 + 8 * half + vv;
            float mx = -3.0e38f;
#pragma unroll
            for (int j = 0; j < 4; ++j) {
                int col = kb * 64 + j * 16 + r;
                float s = sc[j][vv] * scale + (col > qrow ? MASK_VAL : 0.f);
                sc[j][vv] = s;
                mx = fmaxf(mx, s);
            }
#pragma unroll
            for (int off = 1; off < 16; off <<= 1)
                mx = fmaxf(mx, __shfl_xor(mx, off, 32));
            float mnew = fmaxf(mstat[vv], mx);
            float corr = expf(mstat[vv] - mnew);
            float rsum = 0.f;
#pragma unroll
            for (int j = 0; j < 4; ++j) {
                float p = expf(sc[j][vv] - mnew);
                sc[j][vv] = p;
                rsum += p;
            }
#pragma unroll
            for (int off = 1; off < 16; off <<= 1)
                rsum += __shfl_xor(rsum, off, 32);
            lstat[vv] = lstat[vv] * corr + rsum;
            mstat[vv] = mnew;
            oacc[0][vv] *= corr;
            oacc[1][vv] *= corr;
            oacc[2][vv] *= corr;
            oacc[3][vv] *= corr;
        }

        // P (C-layout) -> LDS (row-major bf16) for use as A operand
#pragma unroll
        for (int j = 0; j < 4; ++j)
#pragma unroll
            for (int vv = 0; vv < 8; ++vv)
                ps[(wv * 16 + 8 * half + vv) * 72 + j * 16 + r] = (bf16_t)sc[j][vv];
        __syncthreads();

        // O += P(16x64) x V(64x64)
#pragma unroll
        for (int kk = 0; kk < 2; ++kk) {
            int prow = wv * 16 + r;
            v16bf ap = pack_bf16(*(uint4*)&ps[prow * 72 + kk * 32 + half * 8],
                                 *(uint4*)&ps[prow * 72 + kk * 32 + 16 + half * 8]);
#pragma unroll
            for (int j = 0; j < 4; ++j) {
                int vrow = j * 16 + r;
                v16bf bv = pack_bf16(*(uint4*)&vts[vrow * 72 + kk * 32 + half * 8],
                                     *(uint4*)&vts[vrow * 72 + kk * 32 + 16 + half * 8]);
                oacc[j] = __builtin_amdgcn_wmma_f32_16x16x32_bf16(
                    false, ap, false, bv, (short)0, oacc[j], false, false);
            }
        }
        __syncthreads();
    }

    // Normalize and store bf16
#pragma unroll
    for (int vv = 0; vv < 8; ++vv) {
        float inv = 1.f / lstat[vv];
        int row = tokBase + wv * 16 + 8 * half + vv;
#pragma unroll
        for (int j = 0; j < 4; ++j)
            o[(size_t)row * GD + h * GHD + j * 16 + r] = (bf16_t)(oacc[j][vv] * inv);
    }
}

// ---------------------------------------------------------------------------
// SwiGLU pointwise: g <- silu(g) * u   (bf16 in/out, in place on g)
// ---------------------------------------------------------------------------
__global__ __launch_bounds__(256) void swiglu_kernel(
    bf16_t* __restrict__ g, const bf16_t* __restrict__ u) {
    size_t idx = (size_t)blockIdx.x * 256 + threadIdx.x;
    float gv = (float)g[idx];
    float uv = (float)u[idx];
    float s = gv / (1.f + expf(-gv));
    g[idx] = (bf16_t)(s * uv);
}

// ---------------------------------------------------------------------------
extern "C" void kernel_launch(void* const* d_in, const int* in_sizes, int n_in,
                              void* d_out, int out_size, void* d_ws, size_t ws_size,
                              hipStream_t stream) {
    const float* h = (const float*)d_in[0];
    const float* cosT = (const float*)d_in[1];
    const float* sinT = (const float*)d_in[2];
    const float* wq = (const float*)d_in[3];
    const float* wk = (const float*)d_in[4];
    const float* wv = (const float*)d_in[5];
    const float* wo = (const float*)d_in[6];
    const float* n1 = (const float*)d_in[7];
    const float* n2 = (const float*)d_in[8];
    const float* gate = (const float*)d_in[9];
    const float* gw = (const float*)d_in[10];
    const float* gb = (const float*)d_in[11];
    const float* uw = (const float*)d_in[12];
    const float* ub = (const float*)d_in[13];
    const float* dw = (const float*)d_in[14];
    const float* db = (const float*)d_in[15];

    char* ws = (char*)d_ws;
    const size_t MD2 = (size_t)GM * GD * 2;   // 8 MB  (bf16 [M,D])
    const size_t MD4 = (size_t)GM * GD * 4;   // 16 MB (f32  [M,D])
    const size_t ME2 = (size_t)GM * GE * 2;   // 32 MB (bf16 [M,E])
    bf16_t* x_bf = (bf16_t*)(ws);
    bf16_t* q_bf = (bf16_t*)(ws + MD2);
    bf16_t* k_bf = (bf16_t*)(ws + 2 * MD2);
    bf16_t* v_bf = (bf16_t*)(ws + 3 * MD2);
    bf16_t* a_bf = (bf16_t*)(ws + 4 * MD2);
    bf16_t* x2_bf = (bf16_t*)(ws + 5 * MD2);
    float* h2_f = (float*)(ws + 6 * MD2);
    bf16_t* g_bf = (bf16_t*)(ws + 6 * MD2 + MD4);
    bf16_t* u_bf = (bf16_t*)(ws + 6 * MD2 + MD4 + ME2);

    // 1) x = rmsnorm(h, n1) -> bf16
    rmsnorm_cast_kernel<<<GM, 256, 0, stream>>>(h, n1, x_bf);

    // 2) q,k,v projections
    dim3 gD(GD / 128, GM / 128);
    gemm_bf16_kernel<0><<<gD, 256, 0, stream>>>(x_bf, wq, GM, GD, GD, q_bf, nullptr,
                                                nullptr, nullptr, nullptr, nullptr);
    gemm_bf16_kernel<0><<<gD, 256, 0, stream>>>(x_bf, wk, GM, GD, GD, k_bf, nullptr,
                                                nullptr, nullptr, nullptr, nullptr);
    gemm_bf16_kernel<0><<<gD, 256, 0, stream>>>(x_bf, wv, GM, GD, GD, v_bf, nullptr,
                                                nullptr, nullptr, nullptr, nullptr);

    // 3) RoPE on q,k
    rope_kernel<<<(GM * GH * 32) / 256, 256, 0, stream>>>(q_bf, k_bf, cosT, sinT);

    // 4) causal attention -> a_bf
    attn_kernel<<<dim3(GS / 128, GH, GB), 256, 0, stream>>>(q_bf, k_bf, v_bf, a_bf);

    // 5) h2 = h + attn @ Wo^T
    gemm_bf16_kernel<1><<<gD, 256, 0, stream>>>(a_bf, wo, GM, GD, GD, nullptr, nullptr,
                                                h2_f, h, nullptr, nullptr);

    // 6) x2 = rmsnorm(h2, n2)
    rmsnorm_cast_kernel<<<GM, 256, 0, stream>>>(h2_f, n2, x2_bf);

    // 7) gate / up projections (+bias)
    dim3 gE(GE / 128, GM / 128);
    gemm_bf16_kernel<0><<<gE, 256, 0, stream>>>(x2_bf, gw, GM, GE, GD, g_bf, gb,
                                                nullptr, nullptr, nullptr, nullptr);
    gemm_bf16_kernel<0><<<gE, 256, 0, stream>>>(x2_bf, uw, GM, GE, GD, u_bf, ub,
                                                nullptr, nullptr, nullptr, nullptr);

    // 8) g <- silu(g) * u
    swiglu_kernel<<<(int)(((size_t)GM * GE) / 256), 256, 0, stream>>>(g_bf, u_bf);

    // 9) out = h2 + sigmoid(gate) * ((g*u) @ dw^T + db)
    gemm_bf16_kernel<2><<<gD, 256, 0, stream>>>(g_bf, dw, GM, GD, GE, nullptr, nullptr,
                                                (float*)d_out, h2_f, gate, db);
}
